// LSTM_81011673137158
// MI455X (gfx1250) — matrix operand order repaired and checked by reading, not compiled, not run on hardware
//
#include <hip/hip_runtime.h>
#include <hip/hip_bf16.h>

typedef __attribute__((ext_vector_type(16))) __bf16 v16bf;
typedef __attribute__((ext_vector_type(8)))  float  v8f;
typedef __attribute__((ext_vector_type(4)))  unsigned int v4u;
typedef __attribute__((ext_vector_type(8)))  int    v8i;
typedef __attribute__((ext_vector_type(4)))  int    v4i;

#define C_DIM   512
#define N_NODES 256
#define B_DIM   8
#define T_DIM   32
#define BN      2048          // B*N columns
#define GATES   4
#define KC_PER_C 16           // 512 / 32 K-chunks
#define RB_PER_C 32           // 512 / 16 row-blocks
#define BTILE_ELEMS (4 * KC_PER_C * 512)   // 32768 bf16 = 64 KiB per block B-tile

__device__ __forceinline__ float sigmoidf_fast(float x) {
    return 1.0f / (1.0f + __expf(-x));
}

// ---------------------------------------------------------------------------
// Pack Wx/Wh (fp32, [g][c_out][c_in]) into bf16 WMMA 16x32 A-fragment order:
//   wP[((g*32 + rb)*16 + kc)*512 + lane*16 + e]
// Lane/element -> (row,koff) mapping (contiguous k order, same as the B packing):
//   lane<16 : row=lane,    koff = e
//   lane>=16: row=lane-16, koff = 16+e
// ---------------------------------------------------------------------------
__global__ void prep_weights(const float* __restrict__ Wx, const float* __restrict__ Wh,
                             __bf16* __restrict__ wxP, __bf16* __restrict__ whP) {
    int idx  = blockIdx.x * blockDim.x + threadIdx.x;   // [0, 2*2^20)
    int s    = idx >> 20;
    int rem  = idx & ((1 << 20) - 1);
    int e    = rem & 15;
    int lane = (rem >> 4) & 31;
    int kc   = (rem >> 9) & 15;
    int rb   = (rem >> 13) & 31;
    int g    = (rem >> 18) & 3;
    int row_l, koff;
    if (lane < 16) { row_l = lane;      koff = e;  }
    else           { row_l = lane - 16; koff = 16 + e; }
    int row = rb * 16 + row_l;
    int f   = kc * 32 + koff;
    const float* src = s ? Wh : Wx;
    float v = src[(g * C_DIM + row) * C_DIM + f];
    (s ? whP : wxP)[rem] = (__bf16)v;
}

// Combined bias bsum = bx + bh; zero c-state and the parity-0 h buffer.
__global__ void prep_state(const float* __restrict__ bx, const float* __restrict__ bh,
                           float* __restrict__ bsum, float* __restrict__ cState,
                           __bf16* __restrict__ hP0) {
    int i = blockIdx.x * blockDim.x + threadIdx.x;      // [0, 2^20)
    if (i < GATES * C_DIM) bsum[i] = bx[i] + bh[i];
    cState[i] = 0.0f;
    hP0[i]    = (__bf16)0.0f;
}

// ---------------------------------------------------------------------------
// Pack x[:, :, :, t] (fp32, [b][f][n][t]) into bf16 WMMA 32x16 B-fragment order:
//   xtP[(jb*16 + kc)*512 + lane*16 + e]
//   lane<16 : col = jb*16 + lane,    koff = e
//   lane>=16: col = jb*16 + lane-16, koff = 16 + e
// ---------------------------------------------------------------------------
__global__ void pack_x(const float* __restrict__ x, __bf16* __restrict__ xtP, int t) {
    int idx  = blockIdx.x * blockDim.x + threadIdx.x;   // [0, 2^20)
    int e    = idx & 15;
    int lane = (idx >> 4) & 31;
    int kc   = (idx >> 9) & 15;
    int jb   = idx >> 13;                               // 0..127
    int j    = jb * 16 + (lane & 15);
    int koff = ((lane >> 4) << 4) + e;
    int f    = kc * 32 + koff;
    int b    = j >> 8;
    int n    = j & 255;
    float v = x[((b * C_DIM + f) * N_NODES + n) * T_DIM + t];
    xtP[idx] = (__bf16)v;
}

// ---------------------------------------------------------------------------
// TDM: 1-D contiguous copy global -> LDS via a Tensor DMA Descriptor (D#).
// n8 = length in 8-byte units (data_size=3). Tensor/tile: dim0=n8, dim1=1.
// Group0: count=1 | lds_addr | global_addr[56:0] | type=2.
// Group1 bit layout per CDNA5 ISA.
// This toolchain exposes the 6-arg builtin: (g0, g1, g2, g3, g4, cpol).
// ---------------------------------------------------------------------------
__device__ __forceinline__ void tdm_load_1d(unsigned int lds_off, const void* gptr,
                                            unsigned int n8) {
    unsigned long long ga = (unsigned long long)(uintptr_t)gptr;
    v4u g0;
    g0[0] = 1u;                                              // count=1 (user D#)
    g0[1] = lds_off;                                         // lds_addr (bytes)
    g0[2] = (unsigned int)(ga & 0xFFFFFFFFu);                // global_addr[31:0]
    g0[3] = (unsigned int)((ga >> 32) & 0x01FFFFFFu)         // global_addr[56:32]
          | (2u << 30);                                      // type = 2 ("image")
    v8i g1;
    g1[0] = (int)(3u << 16);                                 // data_size=3 (8B), mask=0
    g1[1] = (int)((n8 & 0xFFFFu) << 16);                     // tensor_dim0[15:0]
    g1[2] = (int)(((n8 >> 16) & 0xFFFFu) | (1u << 16));      // tensor_dim0[31:16], tensor_dim1[15:0]=1
    g1[3] = (int)((n8 & 0xFFFFu) << 16);                     // tensor_dim1[31:16]=0, tile_dim0=n8
    g1[4] = 1;                                               // tile_dim1=1, tile_dim2=0
    g1[5] = (int)n8;                                         // tensor_dim0_stride[31:0]
    g1[6] = 0;                                               // stride hi / dim1_stride lo
    g1[7] = 0;
    v4i z4 = {0, 0, 0, 0};
    v8i z8 = {0, 0, 0, 0, 0, 0, 0, 0};
    __builtin_amdgcn_tensor_load_to_lds(g0, g1, z4, z4, z8, 0);
}

__device__ __forceinline__ unsigned int lds_offset_u32(const void* p) {
    // Generic LDS pointers carry the byte offset in the low 32 bits.
    return (unsigned int)(uintptr_t)p;
}

// ---------------------------------------------------------------------------
// One GEMM phase: acc[g][ns] += A(g, rb-slab) x B(LDS tile), K = 512.
// Fully unrolled; each K-chunk's fragments are fresh SSA values so the
// scheduler pipelines loads across WMMAs without mov/v_nop hazard padding.
// B fragments come from the TDM-staged LDS tile (ds_load_b128 pairs).
// ---------------------------------------------------------------------------
__device__ __forceinline__ void gemm_phase(const __bf16* __restrict__ Ap,
                                           const __bf16* lds,     // block B tile
                                           v8f (&acc)[GATES][2],
                                           int lane, int rb, int nh) {
    const int jb0 = nh * 2;                 // wave's first local 16-col block (0..3)
#pragma unroll
    for (int kc = 0; kc < KC_PER_C; ++kc) {
        v16bf b0 = ((const v16bf*)(lds + (((jb0    ) * KC_PER_C + kc) << 9)))[lane];
        v16bf b1 = ((const v16bf*)(lds + (((jb0 + 1) * KC_PER_C + kc) << 9)))[lane];
        if (kc + 2 < KC_PER_C) {
            __builtin_prefetch((const void*)(Ap + ((size_t)(rb * KC_PER_C + kc + 2) << 9)), 0, 1);
        }
#pragma unroll
        for (int g = 0; g < GATES; ++g) {
            v16bf a = ((const v16bf*)(Ap + ((size_t)((g * RB_PER_C + rb) * KC_PER_C + kc) << 9)))[lane];
            acc[g][0] = __builtin_amdgcn_wmma_f32_16x16x32_bf16(false, a, false, b0,
                                                                (short)0, acc[g][0], false, false);
            acc[g][1] = __builtin_amdgcn_wmma_f32_16x16x32_bf16(false, a, false, b1,
                                                                (short)0, acc[g][1], false, false);
        }
    }
}

// ---------------------------------------------------------------------------
// One LSTM timestep: z[g] = Wx[g]*x_t + Wh[g]*h + (bx+bh), then gate math.
// Block = 8 waves; wave w: mi=w>>1 (16-row slab), nh=w&1 (32 of 64 cols).
// Each wave holds all 4 gate accumulators for its slab -> gate nonlinearity
// is purely in-register. B tiles (x_t and h) are contiguous 64 KiB regions in
// fragment order -> staged into LDS by the Tensor Data Mover (one descriptor
// each), waited with s_wait_tensorcnt.
// ---------------------------------------------------------------------------
__global__ void __launch_bounds__(256)
lstm_step(const __bf16* __restrict__ xtP,
          const __bf16* __restrict__ wxP, const __bf16* __restrict__ whP,
          const float* __restrict__ bsum,
          float* __restrict__ cState,
          const __bf16* __restrict__ hRead, __bf16* __restrict__ hWrite,
          float* __restrict__ out, int t)
{
    __shared__ __align__(16) __bf16 ldsB[2][BTILE_ELEMS];   // 2 x 64 KiB

    const int lane = threadIdx.x & 31;
    const int w    = threadIdx.x >> 5;
    const int mi   = w >> 1;
    const int nh   = w & 1;
    const int rb   = blockIdx.y * 4 + mi;        // 16-row channel block, 0..31
    const int jbB  = blockIdx.x * 4;             // block's first 16-col block

    // Wave 0 kicks off both TDM tile loads (x_t tile and h tile), then waits
    // on TENSORcnt; the workgroup barrier releases the other waves.
    if (w == 0) {
        tdm_load_1d(lds_offset_u32(&ldsB[0][0]),
                    xtP   + (size_t)jbB * KC_PER_C * 512, BTILE_ELEMS / 4);
        tdm_load_1d(lds_offset_u32(&ldsB[1][0]),
                    hRead + (size_t)jbB * KC_PER_C * 512, BTILE_ELEMS / 4);
        __builtin_amdgcn_s_wait_tensorcnt(0);
    }
    __syncthreads();

    // Bias init: accumulator element r maps to row rb*16 + (lane<16 ? r : 8+r),
    // so an aligned v8f load of 8 consecutive bias floats seeds C exactly.
    v8f acc[GATES][2];
#pragma unroll
    for (int g = 0; g < GATES; ++g) {
        const float* bp = bsum + g * C_DIM + rb * 16 + ((lane < 16) ? 0 : 8);
        v8f bias = *(const v8f*)bp;
        acc[g][0] = bias;
        acc[g][1] = bias;
    }

    gemm_phase(wxP, &ldsB[0][0], acc, lane, rb, nh);   // input projection
    gemm_phase(whP, &ldsB[1][0], acc, lane, rb, nh);   // recurrent projection

    // In-register LSTM epilogue.
    const int rowHalf = (lane < 16) ? 0 : 8;
    const int jl      = lane & 15;
#pragma unroll
    for (int ns = 0; ns < 2; ++ns) {
        const int jCol = (jbB + nh * 2 + ns) * 16 + jl;
        const int b    = jCol >> 8;
        const int n    = jCol & 255;
#pragma unroll
        for (int r = 0; r < 8; ++r) {
            const int c = rb * 16 + rowHalf + r;
            float ig = sigmoidf_fast(acc[0][ns][r]);
            float fg = sigmoidf_fast(acc[1][ns][r]);
            float og = sigmoidf_fast(acc[2][ns][r]);
            float gg = tanhf(acc[3][ns][r]);
            const int idxC = c * BN + jCol;
            float cNew = fg * cState[idxC] + ig * gg;
            float hNew = og * tanhf(cNew);
            cState[idxC] = cNew;
            out[((b * C_DIM + c) * N_NODES + n) * T_DIM + t] = hNew;
            // Store h directly in next step's B-fragment order (bf16).
            int kcH   = c >> 5;
            int koff  = c & 31;
            int laneH = (koff < 16) ? jl : (16 + jl);
            int eH    = koff & 15;
            hWrite[(size_t)((jCol >> 4) * KC_PER_C + kcH) * 512 + laneH * 16 + eH] = (__bf16)hNew;
        }
    }
}

// ---------------------------------------------------------------------------
extern "C" void kernel_launch(void* const* d_in, const int* in_sizes, int n_in,
                              void* d_out, int out_size, void* d_ws, size_t ws_size,
                              hipStream_t stream) {
    const float* x  = (const float*)d_in[0];
    const float* Wx = (const float*)d_in[1];
    const float* bx = (const float*)d_in[2];
    const float* Wh = (const float*)d_in[3];
    const float* bh = (const float*)d_in[4];
    float* out = (float*)d_out;

    char* ws = (char*)d_ws;
    // Workspace layout (~14.1 MiB total):
    __bf16* wxP = (__bf16*)(ws);                                      // 2 MiB
    __bf16* whP = (__bf16*)(ws + (1u << 21));                         // 2 MiB
    float*  bs  = (float*) (ws + (1u << 22));                         // 8 KiB
    float*  cS  = (float*) (ws + (1u << 22) + (1u << 16));            // 4 MiB
    __bf16* hP0 = (__bf16*)(ws + (1u << 22) + (1u << 16) + (1u << 22));              // 2 MiB
    __bf16* hP1 = (__bf16*)(ws + (1u << 22) + (1u << 16) + (1u << 22) + (1u << 21)); // 2 MiB
    __bf16* xtP = (__bf16*)(ws + (1u << 22) + (1u << 16) + (1u << 22) + (2u << 21)); // 2 MiB

    prep_weights<<<8192, 256, 0, stream>>>(Wx, Wh, wxP, whP);
    prep_state  <<<4096, 256, 0, stream>>>(bx, bh, bs, cS, hP0);

    dim3 grid(32, 8);  // 32 column tiles (64 cols) x 8 channel tiles (64 rows)
    for (int t = 0; t < T_DIM; ++t) {
        pack_x<<<4096, 256, 0, stream>>>(x, xtP, t);
        const __bf16* hr = (t & 1) ? hP1 : hP0;
        __bf16*       hw = (t & 1) ? hP0 : hP1;
        lstm_step<<<grid, 256, 0, stream>>>(xtP, wxP, whP, bs, cS, hr, hw, out, t);
    }
}
